// LeafBlockAttention_82635170775520
// MI455X (gfx1250) — matrix operand order repaired and checked by reading, not compiled, not run on hardware
//
#include <hip/hip_runtime.h>

// ---------------- problem constants ----------------
#define BB   4
#define NBN  256
#define TT   33
#define CC   512
#define HH   8
#define HD   64
#define NT   (BB * NBN * TT)     // 33792 tokens
#define SCALE 0.125f

typedef __attribute__((ext_vector_type(16))) __bf16 v16bf;
typedef __attribute__((ext_vector_type(8)))  float  v8f;

union FragU { unsigned int u[8]; v16bf v; };

__device__ __forceinline__ unsigned short f32_to_bf16(float f) {
    unsigned int b = __builtin_bit_cast(unsigned int, f);
    unsigned int r = b + 0x7FFFu + ((b >> 16) & 1u);
    return (unsigned short)(r >> 16);
}

// K index for 16-bit A-matrix (16x32) fragment, ISA 7.12.2 layout.
__device__ __forceinline__ int kA(int j, int half) {
    return (j < 4) ? (half * 8 + 2 * j) : (16 + half * 8 + 2 * (j - 4));
}
// K index for 16-bit B-matrix (32x16) fragment, same layout as A.
__device__ __forceinline__ int kB(int j, int half) {
    return (j < 4) ? (half * 8 + 2 * j) : (16 + half * 8 + 2 * (j - 4));
}

__device__ __forceinline__ v8f wmma_bf16(v16bf a, v16bf b, v8f c) {
    return __builtin_amdgcn_wmma_f32_16x16x32_bf16(
        false, a, false, b, (short)0, c, false, false);
}

// Load 16x32 bf16 A-fragment from row-major bf16 memory (pairs are contiguous).
__device__ __forceinline__ v16bf load_fragA(const unsigned short* __restrict__ base,
                                            int row0, int ld, int kbase, int lane) {
    int m = lane & 15, half = lane >> 4;
    const unsigned short* p = base + (size_t)(row0 + m) * ld + kbase;
    FragU f;
#pragma unroll
    for (int j = 0; j < 8; ++j)
        f.u[j] = *(const unsigned int*)(p + kA(j, half));
    return f.v;
}

// Load 32x16 bf16 B-fragment; Bm[k][n] = W[n][k], W row-major (pairs contiguous in k).
__device__ __forceinline__ v16bf load_fragB(const unsigned short* __restrict__ base,
                                            int row0, int ld, int kbase, int lane) {
    int n = lane & 15, half = lane >> 4;
    const unsigned short* p = base + (size_t)(row0 + n) * ld + kbase;
    FragU f;
#pragma unroll
    for (int j = 0; j < 8; ++j)
        f.u[j] = *(const unsigned int*)(p + kB(j, half));
    return f.v;
}

// ---------------- stage 0: f32 -> bf16 cast ----------------
__global__ void cast_f32_bf16(const float* __restrict__ src,
                              unsigned short* __restrict__ dst, int n) {
    int i = blockIdx.x * blockDim.x + threadIdx.x;
    if (i < n) dst[i] = f32_to_bf16(src[i]);
}

// ---------------- stages 1 & 3: C = A @ W^T + bias  (bf16 in, WMMA f32 acc) ---
// A: (M x K) bf16 row-major, W: (N x K) bf16 row-major.
// Block = 128x128 (4 waves, 2x2), wave = 64x64 (4x4 WMMA tiles): 16 WMMA / 8 frags.
template <bool OUT_F32>
__global__ __launch_bounds__(128)
void gemm_bf16_nt(const unsigned short* __restrict__ A,
                  const unsigned short* __restrict__ W,
                  const float* __restrict__ bias,
                  void* __restrict__ Cout, int N, int K) {
    const int lane = threadIdx.x & 31;
    const int wave = threadIdx.x >> 5;
    const int rowW = blockIdx.x * 128 + (wave >> 1) * 64;
    const int colW = blockIdx.y * 128 + (wave & 1) * 64;

    v8f acc[4][4];
#pragma unroll
    for (int i = 0; i < 4; ++i)
#pragma unroll
        for (int j = 0; j < 4; ++j) acc[i][j] = v8f{};

    for (int kb = 0; kb < K; kb += 32) {
        v16bf a[4], b[4];
#pragma unroll
        for (int i = 0; i < 4; ++i) a[i] = load_fragA(A, rowW + i * 16, K, kb, lane);
#pragma unroll
        for (int j = 0; j < 4; ++j) b[j] = load_fragB(W, colW + j * 16, K, kb, lane);
#pragma unroll
        for (int i = 0; i < 4; ++i)
#pragma unroll
            for (int j = 0; j < 4; ++j)
                acc[i][j] = wmma_bf16(a[i], b[j], acc[i][j]);
    }

    const int half = lane >> 4;
#pragma unroll
    for (int i = 0; i < 4; ++i) {
#pragma unroll
        for (int j = 0; j < 4; ++j) {
            int n = colW + j * 16 + (lane & 15);
            float bv = bias[n];
#pragma unroll
            for (int r = 0; r < 8; ++r) {
                int m = rowW + i * 16 + r + 8 * half;
                float val = acc[i][j][r] + bv;
                if (OUT_F32)
                    ((float*)Cout)[(size_t)m * N + n] = val;
                else
                    ((unsigned short*)Cout)[(size_t)m * N + n] = f32_to_bf16(val);
            }
        }
    }
}

// ---------------- stage 2: fused block attention ----------------
// One workgroup per (b,nb) block. Waves 0..7 = heads; wave 8 = async v-stager +
// softmax helper. 288 threads (9 waves).
__global__ __launch_bounds__(288)
void leaf_attn(const unsigned short* __restrict__ qkv,   // (NT, 1536) bf16: q|k|v
               const int*   __restrict__ attn_mask,       // (NB,T,T)
               const float* __restrict__ edge,            // (NB,T,T,4)
               const float* __restrict__ Wgate,           // (H,4)
               const float* __restrict__ bgate,           // (H)
               unsigned short* __restrict__ attnO) {      // (NT, 512) bf16
    __shared__ float s_biasP[TT * TT];                    // bias ch3, diag->1
    __shared__ float s_mask [TT * TT];                    // 0/1
    __shared__ unsigned int s_un[12288];                  // 48KB union: scores f32 / cw bf16
    __shared__ __align__(16) unsigned short s_v[TT * CC]; // 33KB: v block (33 x 512)

    const int blk = blockIdx.x;          // b*NB + nb
    const int nb  = blk % NBN;
    const int token0 = blk * TT;
    const int tid  = threadIdx.x;
    const int wave = tid >> 5;
    const int lane = tid & 31;
    const int half = lane >> 4;

    // ---- phase -1: wave 8 bulk-stages v (33 x 1024B) into LDS with async copies.
    // 66 x global_load_async_to_lds_b128 (32 lanes x 16B = 512B each), ASYNCcnt-tracked.
    if (wave == 8) {
        const unsigned short* gbase = qkv + (size_t)token0 * 1536;   // uniform SGPR base
        unsigned ldsbase = (unsigned)(unsigned long long)(&s_v[0]);  // LDS offset = low 32 bits
#pragma unroll 1
        for (int c = 0; c < 66; ++c) {
            int off   = c * 512 + lane * 16;        // byte offset in packed v region
            int t     = off >> 10;                  // token row (1024B per row)
            int inrow = off & 1023;
            unsigned voff = (unsigned)(t * 3072 + 2048 + inrow); // +2048B = v section
            unsigned laddr = ldsbase + (unsigned)off;
            asm volatile("global_load_async_to_lds_b128 %0, %1, %2"
                         :: "v"(laddr), "v"(voff), "s"(gbase)
                         : "memory");
        }
    }

    // ---- phase 0: stage mask + score-bias into LDS ----
    for (int idx = tid; idx < TT * TT; idx += 288) {
        int q = idx / TT, k = idx % TT;
        int mbase = (nb * TT + q) * TT + k;
        s_mask[idx]  = (attn_mask[mbase] != 0) ? 1.0f : 0.0f;
        s_biasP[idx] = (q == k) ? 1.0f : edge[mbase * 4 + 3];
    }
    __syncthreads();

    // ---- phase 1: scores = (q @ k^T) * SCALE, padded 48x48, per head ----
    float* sc = (float*)s_un;            // view: sc[(h*33+m)*36 + n]
    if (wave < 8) {
        const int h = wave;
        const unsigned short* qb = qkv + (size_t)token0 * 1536 + h * HD;
        const unsigned short* kb = qb + 512;
        v8f sacc[3][3];
#pragma unroll
        for (int i = 0; i < 3; ++i)
#pragma unroll
            for (int j = 0; j < 3; ++j) sacc[i][j] = v8f{};

#pragma unroll
        for (int kk = 0; kk < HD; kk += 32) {
            v16bf aq[3], bk[3];
#pragma unroll
            for (int mt = 0; mt < 3; ++mt) {           // A: q rows (clamped)
                int m = mt * 16 + (lane & 15); if (m > 32) m = 32;
                FragU f;
#pragma unroll
                for (int j = 0; j < 8; ++j)
                    f.u[j] = *(const unsigned int*)(qb + (size_t)m * 1536 + kk + kA(j, half));
                aq[mt] = f.v;
            }
#pragma unroll
            for (int nt = 0; nt < 3; ++nt) {           // B: k rows (clamped), Bm[d][n]=k[n][d]
                int n = nt * 16 + (lane & 15); if (n > 32) n = 32;
                FragU f;
#pragma unroll
                for (int j = 0; j < 8; ++j)
                    f.u[j] = *(const unsigned int*)(kb + (size_t)n * 1536 + kk + kB(j, half));
                bk[nt] = f.v;
            }
#pragma unroll
            for (int mt = 0; mt < 3; ++mt)
#pragma unroll
                for (int nt = 0; nt < 3; ++nt)
                    sacc[mt][nt] = wmma_bf16(aq[mt], bk[nt], sacc[mt][nt]);
        }
#pragma unroll
        for (int mt = 0; mt < 3; ++mt)
#pragma unroll
            for (int nt = 0; nt < 3; ++nt)
#pragma unroll
                for (int r = 0; r < 8; ++r) {
                    int m = mt * 16 + r + 8 * half;
                    int n = nt * 16 + (lane & 15);
                    if (m < TT && n < TT)
                        sc[(h * TT + m) * 36 + n] = sacc[mt][nt][r] * SCALE;
                }
    }
    __syncthreads();

    // ---- phase 2a: softmax + gates, one row (h,q) per thread, in registers ----
    const int row = tid;
    const bool valid = row < HH * TT;
    const int h2 = row / TT, qr = row % TT;
    unsigned int cwpk[32];
#pragma unroll
    for (int j = 0; j < 32; ++j) cwpk[j] = 0u;

    if (valid) {
        float wg0 = Wgate[h2 * 4 + 0], wg1 = Wgate[h2 * 4 + 1];
        float wg2 = Wgate[h2 * 4 + 2], wg3 = Wgate[h2 * 4 + 3];
        float bg  = bgate[h2];
        float sv[TT];
        float maxv = -3.0e38f;
#pragma unroll
        for (int k = 0; k < TT; ++k) {
            float s = sc[(h2 * TT + qr) * 36 + k] + s_biasP[qr * TT + k];
            sv[k] = s;
            if (s_mask[qr * TT + k] != 0.0f && s > maxv) maxv = s;
        }
        float sum = 0.0f;
#pragma unroll
        for (int k = 0; k < TT; ++k) {
            float e = (s_mask[qr * TT + k] != 0.0f) ? __expf(sv[k] - maxv) : 0.0f;
            sv[k] = e; sum += e;
        }
        float inv = 1.0f / sum;
#pragma unroll
        for (int k = 0; k < TT; ++k) {
            float g;
            if (qr == k) g = wg3 + bg;                 // diag bias = (0,0,0,1)
            else {
                const float* e4 = edge + ((size_t)(nb * TT + qr) * TT + k) * 4;
                g = e4[0] * wg0 + e4[1] * wg1 + e4[2] * wg2 + e4[3] * wg3 + bg;
            }
            float cwv = sv[k] * inv + ((s_mask[qr * TT + k] != 0.0f) ? g : 0.0f);
            unsigned short b = f32_to_bf16(cwv);
            cwpk[k >> 1] |= ((unsigned int)b) << ((k & 1) * 16);
        }
    }
    __syncthreads();   // everyone done READING scores before cw overwrites the union

    // ---- phase 2b: write cw rows (bf16, k padded to 64) + zero pad rows ----
    if (valid) {
        unsigned int* cwrow = s_un + (h2 * 48 + qr) * 32;
#pragma unroll
        for (int j = 0; j < 32; ++j) cwrow[j] = cwpk[j];
    }
    for (int w = tid; w < 8 * 15 * 32; w += 288) {     // rows 33..47 per head -> 0
        int hz = w / (15 * 32), rem = w % (15 * 32);
        s_un[(hz * 48 + 33 + rem / 32) * 32 + (rem % 32)] = 0u;
    }
    if (wave == 8)                                     // v staging must be complete
        asm volatile("s_wait_asynccnt 0x0" ::: "memory");
    __syncthreads();

    // ---- phase 3: out = cw @ v  (48x64, K=64 padded, cw zero beyond 33) ----
    if (wave < 8) {
        const int h = wave;
        v8f oacc[3][4];
#pragma unroll
        for (int i = 0; i < 3; ++i)
#pragma unroll
            for (int j = 0; j < 4; ++j) oacc[i][j] = v8f{};

#pragma unroll
        for (int kk = 0; kk < 64; kk += 32) {
            v16bf af[3], bf[4];
#pragma unroll
            for (int mt = 0; mt < 3; ++mt) {           // A = cw from LDS (ds loads)
                int m = mt * 16 + (lane & 15);
                FragU f;
#pragma unroll
                for (int j = 0; j < 8; ++j)
                    f.u[j] = s_un[(h * 48 + m) * 32 + ((kk + kA(j, half)) >> 1)];
                af[mt] = f.v;
            }
#pragma unroll
            for (int nt = 0; nt < 4; ++nt) {           // B = v from LDS (staged async)
                int col = h * HD + nt * 16 + (lane & 15);
                FragU f;
#pragma unroll
                for (int j = 0; j < 8; ++j) {
                    int kp = kk + kB(j, half);
                    int t0 = kp     > 32 ? 32 : kp;
                    int t1 = kp + 1 > 32 ? 32 : kp + 1;
                    unsigned int lo = s_v[t0 * CC + col];
                    unsigned int hi = s_v[t1 * CC + col];
                    f.u[j] = lo | (hi << 16);
                }
                bf[nt] = f.v;
            }
#pragma unroll
            for (int mt = 0; mt < 3; ++mt)
#pragma unroll
                for (int nt = 0; nt < 4; ++nt)
                    oacc[mt][nt] = wmma_bf16(af[mt], bf[nt], oacc[mt][nt]);
        }
#pragma unroll
        for (int mt = 0; mt < 3; ++mt)
#pragma unroll
            for (int nt = 0; nt < 4; ++nt)
#pragma unroll
                for (int r = 0; r < 8; ++r) {
                    int m = mt * 16 + r + 8 * half;
                    if (m < TT) {
                        int col = h * HD + nt * 16 + (lane & 15);
                        attnO[(size_t)(token0 + m) * CC + col] = f32_to_bf16(oacc[mt][nt][r]);
                    }
                }
    }
}

// ---------------- host launcher ----------------
extern "C" void kernel_launch(void* const* d_in, const int* in_sizes, int n_in,
                              void* d_out, int out_size, void* d_ws, size_t ws_size,
                              hipStream_t stream) {
    const float* x     = (const float*)d_in[0];
    const int*   amask = (const int*)  d_in[1];
    const float* edge  = (const float*)d_in[2];
    const float* Wqkv  = (const float*)d_in[3];
    const float* bqkv  = (const float*)d_in[4];
    const float* Wproj = (const float*)d_in[5];
    const float* bproj = (const float*)d_in[6];
    const float* Wgate = (const float*)d_in[7];
    const float* bgate = (const float*)d_in[8];
    float* out = (float*)d_out;

    char* ws = (char*)d_ws;
    unsigned short* x_bf     = (unsigned short*)(ws);
    unsigned short* Wqkv_bf  = (unsigned short*)(ws + 34603008);
    unsigned short* Wproj_bf = (unsigned short*)(ws + 36175872);
    unsigned short* qkv_bf   = (unsigned short*)(ws + 36700160);
    unsigned short* attnO_bf = (unsigned short*)(ws + 140509184);

    // stage 0: casts
    cast_f32_bf16<<<(NT * CC + 255) / 256, 256, 0, stream>>>(x, x_bf, NT * CC);
    cast_f32_bf16<<<(1536 * 512 + 255) / 256, 256, 0, stream>>>(Wqkv, Wqkv_bf, 1536 * 512);
    cast_f32_bf16<<<(512 * 512 + 255) / 256, 256, 0, stream>>>(Wproj, Wproj_bf, 512 * 512);

    // stage 1: qkv = x @ Wqkv^T + bqkv   (M=33792, N=1536, K=512) -> bf16
    gemm_bf16_nt<false><<<dim3(NT / 128, 1536 / 128), 128, 0, stream>>>(
        x_bf, Wqkv_bf, bqkv, (void*)qkv_bf, 1536, 512);

    // stage 2: fused block attention -> attnO bf16
    leaf_attn<<<BB * NBN, 288, 0, stream>>>(qkv_bf, amask, edge, Wgate, bgate, attnO_bf);

    // stage 3: out = attnO @ Wproj^T + bproj -> f32
    gemm_bf16_nt<true><<<dim3(NT / 128, 512 / 128), 128, 0, stream>>>(
        attnO_bf, Wproj_bf, bproj, (void*)out, 512, 512);
}
